// GatedLinearAttention_11776800326035
// MI455X (gfx1250) — matrix operand order repaired and checked by reading, not compiled, not run on hardware
//
#include <hip/hip_runtime.h>
#include <hip/hip_bf16.h>
#include <stdint.h>

// ---------------------------------------------------------------------------
// GLA forward for MI455X (gfx1250, wave32, WMMA bf16 16x16x32, f32 accum,
// double-buffered async global->LDS staging for both GEMM tiles)
// ---------------------------------------------------------------------------
static constexpr int cHID = 2048;
static constexpr int cH   = 8;
static constexpr int cDK  = 128;
static constexpr int cDV  = 256;
static constexpr int cKD  = cH * cDK;   // 1024
static constexpr int cVD  = cH * cDV;   // 2048
static constexpr int cCH  = 64;         // chunk
static constexpr int cSC  = 16;         // subchunk
static constexpr int cNS  = cCH / cSC;  // 4
static constexpr int cLRD = 16;

typedef __bf16 bf16_t;
typedef bf16_t v16bf __attribute__((ext_vector_type(16)));
typedef float  v8f   __attribute__((ext_vector_type(8)));

__device__ __forceinline__ unsigned short f2bf(float f) {
  unsigned u = __float_as_uint(f);
  u += 0x7FFFu + ((u >> 16) & 1u);          // round-to-nearest-even
  return (unsigned short)(u >> 16);
}
__device__ __forceinline__ float bf2f(unsigned short s) {
  return __uint_as_float(((unsigned)s) << 16);
}
__device__ __forceinline__ unsigned pk2(float lo, float hi) {
  return (unsigned)f2bf(lo) | ((unsigned)f2bf(hi) << 16);
}
__device__ __forceinline__ v8f vzero8() {
  v8f z = {0.f, 0.f, 0.f, 0.f, 0.f, 0.f, 0.f, 0.f};
  return z;
}
__device__ __forceinline__ v8f wmma_bf16(v16bf a, v16bf b, v8f c) {
  // D = A(16x32 bf16) * B(32x16 bf16) + C(16x16 f32)
  return __builtin_amdgcn_wmma_f32_16x16x32_bf16(false, a, false, b,
                                                 (short)0, c, false, false);
}

// Async DMA: copy 16B from each lane's global address into each lane's LDS
// offset (GLOBAL_LOAD_ASYNC_TO_LDS_B128, tracked by ASYNCcnt, in-order).
__device__ __forceinline__ void async_b128(unsigned lds_off, const void* gptr) {
  asm volatile("global_load_async_to_lds_b128 %0, %1, off"
               :: "v"(lds_off), "v"((unsigned long long)(uintptr_t)gptr)
               : "memory");
}
__device__ __forceinline__ void wait_async0() {
  asm volatile("s_wait_asynccnt 0x0" ::: "memory");
}
__device__ __forceinline__ void wait_async4() {  // oldest tile done; newest 4 in flight
  asm volatile("s_wait_asynccnt 0x4" ::: "memory");
}
__device__ __forceinline__ unsigned lds_off32(const void* p) {
  return (unsigned)(uintptr_t)p;  // generic LDS addr = {aperture, offset32}
}

union FragU { uint4 u[2]; v16bf v; unsigned short s[16]; };

// A-matrix 16x32 bf16 fragment: lane holds K in {g*16..g*16+15}
// with g = lane/16 (row = base + lane%16), matching ldfragB.
__device__ __forceinline__ v16bf ldfragA(const unsigned short* row, int lane) {
  const int off = (lane >> 4) * 16;
  FragU f;
  f.u[0] = *(const uint4*)(row + off);
  f.u[1] = *(const uint4*)(row + off + 8);
  return f.v;
}
// B-matrix 32x16 bf16 fragment from [n][k] storage: lane holds K in
// {g*16..g*16+15}, g = lane/16 (row n = base + lane%16). One 32B chunk.
__device__ __forceinline__ v16bf ldfragB(const unsigned short* row, int lane) {
  const int off = (lane >> 4) * 16;
  FragU f;
  f.u[0] = *(const uint4*)(row + off);
  f.u[1] = *(const uint4*)(row + off + 8);
  return f.v;
}

// ---------------------------------------------------------------------------
// hidden_states f32 -> bf16 (one pass)
// ---------------------------------------------------------------------------
__global__ __launch_bounds__(256) void cvt_bf16_kernel(const float* __restrict__ in,
                                                       unsigned short* __restrict__ out) {
  const size_t i = (size_t)blockIdx.x * 256 + threadIdx.x;  // float4 units
  const float4 f = ((const float4*)in)[i];
  uint2 p;
  p.x = pk2(f.x, f.y);
  p.y = pk2(f.z, f.w);
  ((uint2*)out)[i] = p;
}

// ---------------------------------------------------------------------------
// Weight transpose+convert: W f32 [K][N] -> WT bf16 [N][K] (LDS-tiled 32x32).
// One-time cost; makes GEMM B staging a pure async-DMA-able bf16 copy.
// ---------------------------------------------------------------------------
__global__ __launch_bounds__(256) void wtrans_kernel(const float* __restrict__ W,
                                                     unsigned short* __restrict__ WT,
                                                     int K, int N) {
  __shared__ unsigned short t[32][33];
  const int k0 = blockIdx.y * 32, n0 = blockIdx.x * 32;
  const int c = threadIdx.x & 31, r0 = threadIdx.x >> 5;
#pragma unroll
  for (int rr = 0; rr < 4; ++rr) {
    const int r = r0 + rr * 8;                       // k_local
    t[c][r] = f2bf(W[(size_t)(k0 + r) * N + n0 + c]);  // t[n_local][k_local]
  }
  __syncthreads();
#pragma unroll
  for (int rr = 0; rr < 4; ++rr) {
    const int r = r0 + rr * 8;                       // n_local
    WT[(size_t)(n0 + r) * K + k0 + c] = t[r][c];
  }
}

// ---------------------------------------------------------------------------
// 128x128-tile GEMM: C[M,N] = A[M,K] @ B[K,N]; bf16 WMMA, f32 accum.
// A bf16 [M][K], BT bf16 [N][K]; both tiles async-DMA'd, double-buffered.
// ---------------------------------------------------------------------------
static constexpr int LDT = 40;  // LDS row stride in bf16 elems (32 + pad)

template <bool C_BF16>
__global__ __launch_bounds__(256) void gemm_kernel(const unsigned short* __restrict__ A,
                                                   const unsigned short* __restrict__ BT,
                                                   void* __restrict__ Cp,
                                                   int M, int N, int K) {
  __shared__ unsigned short sA[2][128 * LDT];
  __shared__ unsigned short sB[2][128 * LDT];
  const int tid  = threadIdx.x;
  const int lane = tid & 31, wid = tid >> 5;
  const int wm = wid >> 1, wn = wid & 1;
  const int lr = lane & 15, lh = lane >> 4;
  const long bm = (long)blockIdx.y * 128, bn = (long)blockIdx.x * 128;

  v8f acc[2][4];
#pragma unroll
  for (int i = 0; i < 2; ++i)
#pragma unroll
    for (int j = 0; j < 4; ++j) acc[i][j] = vzero8();

  // staging: thread -> (row, 16-elem col chunk); 32B per thread per tile
  const int sr = tid >> 1, sc = (tid & 1) * 16;
  const unsigned dA[2] = {lds_off32(&sA[0][sr * LDT + sc]),
                          lds_off32(&sA[1][sr * LDT + sc])};
  const unsigned dB[2] = {lds_off32(&sB[0][sr * LDT + sc]),
                          lds_off32(&sB[1][sr * LDT + sc])};
  const unsigned short* srcA = A  + (size_t)(bm + sr) * K + sc;
  const unsigned short* srcB = BT + (size_t)(bn + sr) * K + sc;

  // prologue: stage k-tile 0 into buffer 0
  async_b128(dA[0], srcA);
  async_b128(dA[0] + 16, srcA + 8);
  async_b128(dB[0], srcB);
  async_b128(dB[0] + 16, srcB + 8);

  int cur = 0;
  for (int kk = 0; kk < K; kk += 32) {
    __syncthreads();  // all waves done reading buffer cur^1 (written next)
    if (kk + 32 < K) {
      const int nxt = cur ^ 1;
      const unsigned short* nA = srcA + kk + 32;
      const unsigned short* nB = srcB + kk + 32;
      async_b128(dA[nxt], nA);
      async_b128(dA[nxt] + 16, nA + 8);
      async_b128(dB[nxt], nB);
      async_b128(dB[nxt] + 16, nB + 8);
      if (kk + 64 < K) __builtin_prefetch(srcA + kk + 64, 0, 1);  // warm L2
      wait_async4();  // in-order: current tile's 4 copies have landed
    } else {
      wait_async0();
    }
    __syncthreads();  // all waves' copies for buffer cur visible

    v16bf af[2], bfr[4];
#pragma unroll
    for (int mf = 0; mf < 2; ++mf)
      af[mf] = ldfragA(&sA[cur][(wm * 32 + mf * 16 + lr) * LDT], lane);
#pragma unroll
    for (int nf = 0; nf < 4; ++nf)
      bfr[nf] = ldfragB(&sB[cur][(wn * 64 + nf * 16 + lr) * LDT], lane);
#pragma unroll
    for (int mf = 0; mf < 2; ++mf)
#pragma unroll
      for (int nf = 0; nf < 4; ++nf)
        acc[mf][nf] = wmma_bf16(af[mf], bfr[nf], acc[mf][nf]);
    cur ^= 1;
  }

  // ---- epilogue: C/D layout lane(L%16)=n, row m = r + 8*(L/16) ----------
#pragma unroll
  for (int mf = 0; mf < 2; ++mf)
#pragma unroll
    for (int nf = 0; nf < 4; ++nf)
#pragma unroll
      for (int r = 0; r < 8; ++r) {
        const long row = bm + wm * 32 + mf * 16 + lh * 8 + r;
        const long col = bn + wn * 64 + nf * 16 + lr;
        if (C_BF16)
          ((unsigned short*)Cp)[(size_t)row * N + col] = f2bf(acc[mf][nf][r]);
        else
          ((float*)Cp)[(size_t)row * N + col] = acc[mf][nf][r];
      }
}

// ---------------------------------------------------------------------------
// Low-rank gate path: gkl = hs @ Wgk1  (K=2048, N=16) — f32 for gate fidelity
// ---------------------------------------------------------------------------
__global__ __launch_bounds__(256) void lr1_kernel(const float* __restrict__ hs,
                                                  const float* __restrict__ W1,
                                                  float* __restrict__ gkl) {
  const int g = blockIdx.x * 256 + threadIdx.x;
  const int m = g >> 4, n = g & 15;
  const float* hr = hs + (size_t)m * cHID;
  float acc = 0.f;
#pragma unroll 4
  for (int i = 0; i < cHID; ++i) acc += hr[i] * W1[i * cLRD + n];
  gkl[g] = acc;
}

// gk = log_sigmoid(gkl @ Wgk2 + b) / 16   (K=16, N=1024)
__global__ __launch_bounds__(256) void lr2_kernel(const float* __restrict__ gkl,
                                                  const float* __restrict__ W2,
                                                  const float* __restrict__ b2,
                                                  float* __restrict__ gkout) {
  const int g = blockIdx.x * 256 + threadIdx.x;
  const int m = g >> 10, n = g & 1023;
  const float* gr = gkl + (size_t)m * cLRD;
  float acc = b2[n];
#pragma unroll
  for (int i = 0; i < cLRD; ++i) acc += gr[i] * W2[i * cKD + n];
  const float ls = fminf(acc, 0.f) - log1pf(expf(-fabsf(acc)));  // log_sigmoid
  gkout[g] = ls * (1.f / 16.f);
}

// ---------------------------------------------------------------------------
// Gate precompute per (b,h,chunk): cumsum + decayed q/k tensors (bf16).
//   qEx[b,t,h,d]  = q * exp(Blast)           (row-major in d)
//   qInn[b,t,h,d] = q * exp(Bc)
//   kInn[b,t,h,d] = k * exp(-Bc)
//   kTex[b,nc,h,d,c] = k * exp(Blast - Bc)   (transposed: contiguous in c)
// ---------------------------------------------------------------------------
__global__ __launch_bounds__(128) void gate_kernel(const float* __restrict__ q,
                                                   const float* __restrict__ k,
                                                   const float* __restrict__ gk,
                                                   unsigned short* __restrict__ qEx,
                                                   unsigned short* __restrict__ qInn,
                                                   unsigned short* __restrict__ kInn,
                                                   unsigned short* __restrict__ kTex,
                                                   float* __restrict__ eBlast,
                                                   int T, int NC) {
  int idx = blockIdx.x;
  const int nc = idx % NC; idx /= NC;
  const int h  = idx % cH; idx /= cH;
  const int b  = idx;
  const int d  = threadIdx.x;
  const size_t colbase = ((size_t)(b * T) + (size_t)nc * cCH) * cKD + h * cDK + d;

  float acc = 0.f;
  for (int c = 0; c < cCH; ++c) acc += gk[colbase + (size_t)c * cKD];
  const float Bl = acc, eb = expf(acc);

  const size_t cb = ((size_t)(b * NC + nc)) * cH + h;
  eBlast[cb * cDK + d] = eb;
  unsigned short* ktr = kTex + (cb * cDK + d) * cCH;

  acc = 0.f;
  for (int c = 0; c < cCH; ++c) {
    const size_t ix = colbase + (size_t)c * cKD;
    acc += gk[ix];
    const float qv = q[ix], kv = k[ix];
    qEx[ix]  = f2bf(qv * eb);
    qInn[ix] = f2bf(qv * expf(acc));
    kInn[ix] = f2bf(kv * expf(-acc));
    ktr[c]   = f2bf(kv * expf(Bl - acc));
  }
}

// v (bf16, [b,t,h,e]) -> vT (bf16, [b,nc,h,e,c]) for WMMA B-fragments.
__global__ __launch_bounds__(256) void vtrans_kernel(const unsigned short* __restrict__ vbf,
                                                     unsigned short* __restrict__ vT,
                                                     int T, int NC) {
  int idx = blockIdx.x;
  const int h  = idx % cH; idx /= cH;
  const int nc = idx % NC; idx /= NC;
  const int b  = idx;
  const int e  = threadIdx.x;
  unsigned short* dst = vT + (((size_t)(b * NC + nc) * cH + h) * cDV + e) * cCH;
  for (int c = 0; c < cCH; ++c)
    dst[c] = vbf[(((size_t)(b * T) + (size_t)nc * cCH + c) * cH + h) * cDV + e];
}

// ---------------------------------------------------------------------------
// Chunked recurrence: grid (es, h, b); S[128 x 64] slice lives in LDS.
//   S <- eBlast * S + k_exp^T @ v ; O_inter = q_exp @ S  (WMMA bf16)
// ---------------------------------------------------------------------------
static constexpr int SP = 68;  // padded S row stride (f32)

__global__ __launch_bounds__(256) void scan_kernel(const unsigned short* __restrict__ kTex,
                                                   const unsigned short* __restrict__ vT,
                                                   const unsigned short* __restrict__ qEx,
                                                   const float* __restrict__ eBlast,
                                                   float* __restrict__ o,
                                                   int T, int NC) {
  __shared__ float S[cDK * SP];
  __shared__ float eB[cDK];
  const int es = blockIdx.x, h = blockIdx.y, b = blockIdx.z;
  const int tid = threadIdx.x, lane = tid & 31, wid = tid >> 5;
  const int lr = lane & 15, lh = lane >> 4;

  for (int i = tid; i < cDK * SP; i += 256) S[i] = 0.f;
  __syncthreads();

  for (int nc = 0; nc < NC; ++nc) {
    const size_t cb = ((size_t)(b * NC + nc)) * cH + h;
    if (tid < cDK) eB[tid] = eBlast[cb * cDK + tid];
    __syncthreads();

    // ---- S update: wave wid owns d-tile dt = wid (16 rows) --------------
    {
      const int dt = wid;
      const unsigned short* kr = kTex + (cb * cDK + dt * 16 + lr) * cCH;
      const v16bf a0 = ldfragA(kr, lane);        // c in [0,32)
      const v16bf a1 = ldfragA(kr + 32, lane);   // c in [32,64)
#pragma unroll
      for (int et = 0; et < 4; ++et) {
        const int e0 = es * 64 + et * 16;
        const unsigned short* vr = vT + (cb * cDV + e0 + lr) * cCH;
        const v16bf b0 = ldfragB(vr, lane);
        const v16bf b1 = ldfragB(vr + 32, lane);
        v8f c;
#pragma unroll
        for (int r = 0; r < 8; ++r) {
          const int row = dt * 16 + lh * 8 + r;
          c[r] = S[row * SP + et * 16 + lr] * eB[row];
        }
        c = wmma_bf16(a0, b0, c);
        c = wmma_bf16(a1, b1, c);
#pragma unroll
        for (int r = 0; r < 8; ++r) {
          const int row = dt * 16 + lh * 8 + r;
          S[row * SP + et * 16 + lr] = c[r];
        }
      }
    }
    __syncthreads();

    // ---- O_inter = q_exp @ S_new ---------------------------------------
    {
      const int ct = wid & 3, eh = wid >> 2;
#pragma unroll
      for (int ei = 0; ei < 2; ++ei) {
        const int et = eh * 2 + ei;
        v8f acc = vzero8();
#pragma unroll
        for (int kk = 0; kk < 4; ++kk) {
          const size_t trow = (size_t)(b * T) + (size_t)nc * cCH + ct * 16 + lr;
          const v16bf a = ldfragA(qEx + (trow * cH + h) * cDK + kk * 32, lane);
          FragU bb;
          const int nloc = et * 16 + lr;
#pragma unroll
          for (int i = 0; i < 16; ++i)
            bb.s[i] = f2bf(S[(kk * 32 + lh * 16 + i) * SP + nloc]);
          acc = wmma_bf16(a, bb.v, acc);
        }
#pragma unroll
        for (int r = 0; r < 8; ++r) {
          const int t = nc * cCH + ct * 16 + lh * 8 + r;
          const int e = es * 64 + et * 16 + lr;
          o[(((size_t)(b * T) + t) * cH + h) * cDV + e] = acc[r];
        }
      }
    }
    __syncthreads();
  }
}

// ---------------------------------------------------------------------------
// Intra (tril) term per (b,nc,subchunk,h): P = (q e^{Bc})(k e^{-Bc})^T,
// tril mask, column-sum, o += colsum * v  (one global f32 atomic per elem).
// ---------------------------------------------------------------------------
__global__ __launch_bounds__(32) void intra_kernel(const unsigned short* __restrict__ qInn,
                                                   const unsigned short* __restrict__ kInn,
                                                   const unsigned short* __restrict__ vbf,
                                                   float* __restrict__ o,
                                                   int T, int NC) {
  int idx = blockIdx.x;
  const int h  = idx % cH;  idx /= cH;
  const int j  = idx % cNS; idx /= cNS;
  const int nc = idx % NC;  idx /= NC;
  const int b  = idx;
  const int lane = threadIdx.x;
  const int lr = lane & 15, lh = lane >> 4;
  const int t0 = nc * cCH + j * cSC;

  v8f acc = vzero8();
#pragma unroll
  for (int kk = 0; kk < 4; ++kk) {
    const size_t row = ((size_t)(b * T) + t0 + lr) * cH + h;
    const v16bf a  = ldfragA(qInn + row * cDK + kk * 32, lane);
    const v16bf bb = ldfragB(kInn + row * cDK + kk * 32, lane);
    acc = wmma_bf16(a, bb, acc);
  }
  // tril mask (keep m >= n) + column sum over m
  float cs = 0.f;
#pragma unroll
  for (int r = 0; r < 8; ++r) {
    const int m = lh * 8 + r;
    if (m >= lr) cs += acc[r];
  }
  cs += __shfl_xor(cs, 16);  // combine the two m-halves; lane holds colsum[lr]

  for (int nn = 0; nn < 16; ++nn) {
    const float csn = __shfl(cs, nn);
    const size_t base = (((size_t)(b * T) + t0 + nn) * cH + h) * cDV;
    for (int e = lane; e < cDV; e += 32)
      atomicAdd(&o[base + e], csn * bf2f(vbf[base + e]));
  }
}

// ---------------------------------------------------------------------------
// x = silu(LayerNorm_{DV}(o + g) * ln_w + ln_b)  ->  bf16 [B*T, VD]
// ---------------------------------------------------------------------------
__global__ __launch_bounds__(256) void post_kernel(const float* __restrict__ o,
                                                   const float* __restrict__ g,
                                                   const float* __restrict__ lnw,
                                                   const float* __restrict__ lnb,
                                                   unsigned short* __restrict__ xbf) {
  __shared__ float red[256];
  const size_t row = blockIdx.x;
  const int e = threadIdx.x;
  const float x = o[row * cDV + e] + g[row * cDV + e];

  red[e] = x;
  __syncthreads();
  for (int s = 128; s > 0; s >>= 1) { if (e < s) red[e] += red[e + s]; __syncthreads(); }
  const float mu = red[0] * (1.f / cDV);
  __syncthreads();

  const float d = x - mu;
  red[e] = d * d;
  __syncthreads();
  for (int s = 128; s > 0; s >>= 1) { if (e < s) red[e] += red[e + s]; __syncthreads(); }
  const float var = red[0] * (1.f / cDV);

  const float y = d * rsqrtf(var + 1e-5f) * lnw[e] + lnb[e];
  const float s = y / (1.f + expf(-y));  // silu
  xbf[row * cDV + e] = f2bf(s);
}

// ---------------------------------------------------------------------------
extern "C" void kernel_launch(void* const* d_in, const int* in_sizes, int n_in,
                              void* d_out, int out_size, void* d_ws, size_t ws_size,
                              hipStream_t stream) {
  const float* hs  = (const float*)d_in[0];
  const float* Wq  = (const float*)d_in[1];
  const float* Wk  = (const float*)d_in[2];
  const float* Wv  = (const float*)d_in[3];
  const float* Wg  = (const float*)d_in[4];
  const float* W1  = (const float*)d_in[5];
  const float* W2  = (const float*)d_in[6];
  const float* b2  = (const float*)d_in[7];
  const float* Wo  = (const float*)d_in[8];
  const float* lnw = (const float*)d_in[9];
  const float* lnb = (const float*)d_in[10];

  const int BT = in_sizes[0] / cHID;  // 4096
  const int Bb = 2;                   // per reference setup
  const int T  = BT / Bb;             // 2048
  const int NC = T / cCH;             // 32

  char* ws = (char*)d_ws;
  size_t off = 0;
  auto alloc = [&](size_t bytes) -> void* {
    void* p = ws + off;
    off = (off + bytes + 255) & ~(size_t)255;
    return p;
  };
  unsigned short* hsbf = (unsigned short*)alloc((size_t)BT * cHID * 2);
  unsigned short* WqT  = (unsigned short*)alloc((size_t)cHID * cKD * 2);
  unsigned short* WkT  = (unsigned short*)alloc((size_t)cHID * cKD * 2);
  unsigned short* WvT  = (unsigned short*)alloc((size_t)cHID * cVD * 2);
  unsigned short* WgT  = (unsigned short*)alloc((size_t)cHID * cVD * 2);
  unsigned short* WoT  = (unsigned short*)alloc((size_t)cVD * cHID * 2);
  float*          q    = (float*)alloc((size_t)BT * cKD * 4);
  float*          kbuf = (float*)alloc((size_t)BT * cKD * 4);
  unsigned short* vbf  = (unsigned short*)alloc((size_t)BT * cVD * 2);
  float*          gbuf = (float*)alloc((size_t)BT * cVD * 4);
  float*          gkl  = (float*)alloc((size_t)BT * cLRD * 4);
  float*          gk   = (float*)alloc((size_t)BT * cKD * 4);
  unsigned short* qEx  = (unsigned short*)alloc((size_t)BT * cKD * 2);
  unsigned short* qInn = (unsigned short*)alloc((size_t)BT * cKD * 2);
  unsigned short* kInn = (unsigned short*)alloc((size_t)BT * cKD * 2);
  unsigned short* kTex = (unsigned short*)alloc((size_t)BT * cKD * 2);
  float*          eBl  = (float*)alloc((size_t)Bb * NC * cH * cDK * 4);
  unsigned short* vT   = (unsigned short*)alloc((size_t)Bb * NC * cH * cDV * cCH * 2);
  float*          obuf = (float*)alloc((size_t)BT * cVD * 4);
  unsigned short* xbf  = (unsigned short*)alloc((size_t)BT * cVD * 2);
  (void)ws_size; (void)n_in; (void)out_size;

  // 0) one-time conversions: hs -> bf16; weights -> bf16 transposed [N][K]
  cvt_bf16_kernel<<<(BT * cHID / 4) / 256, 256, 0, stream>>>(hs, hsbf);
  wtrans_kernel<<<dim3(cKD / 32, cHID / 32), 256, 0, stream>>>(Wq, WqT, cHID, cKD);
  wtrans_kernel<<<dim3(cKD / 32, cHID / 32), 256, 0, stream>>>(Wk, WkT, cHID, cKD);
  wtrans_kernel<<<dim3(cVD / 32, cHID / 32), 256, 0, stream>>>(Wv, WvT, cHID, cVD);
  wtrans_kernel<<<dim3(cVD / 32, cHID / 32), 256, 0, stream>>>(Wg, WgT, cHID, cVD);
  wtrans_kernel<<<dim3(cHID / 32, cVD / 32), 256, 0, stream>>>(Wo, WoT, cVD, cHID);

  // 1) projections (bf16 WMMA GEMMs, double-buffered async staging)
  gemm_kernel<false><<<dim3(cKD / 128, BT / 128), 256, 0, stream>>>(hsbf, WqT, q,    BT, cKD, cHID);
  gemm_kernel<false><<<dim3(cKD / 128, BT / 128), 256, 0, stream>>>(hsbf, WkT, kbuf, BT, cKD, cHID);
  gemm_kernel<true ><<<dim3(cVD / 128, BT / 128), 256, 0, stream>>>(hsbf, WvT, vbf,  BT, cVD, cHID);
  gemm_kernel<false><<<dim3(cVD / 128, BT / 128), 256, 0, stream>>>(hsbf, WgT, gbuf, BT, cVD, cHID);

  // 2) low-rank gate path (f32 for gate fidelity)
  lr1_kernel<<<(BT * cLRD) / 256, 256, 0, stream>>>(hs, W1, gkl);
  lr2_kernel<<<(BT * cKD) / 256, 256, 0, stream>>>(gkl, W2, b2, gk);

  // 3) gate precompute + v transpose
  gate_kernel<<<Bb * cH * NC, 128, 0, stream>>>(q, kbuf, gk, qEx, qInn, kInn, kTex, eBl, T, NC);
  vtrans_kernel<<<Bb * NC * cH, 256, 0, stream>>>(vbf, vT, T, NC);

  // 4) chunked recurrence (O_inter)
  scan_kernel<<<dim3(cDV / 64, cH, Bb), 256, 0, stream>>>(kTex, vT, qEx, eBl, obuf, T, NC);

  // 5) intra-chunk tril term (atomic add into o)
  intra_kernel<<<Bb * NC * cNS * cH, 32, 0, stream>>>(qInn, kInn, vbf, obuf, T, NC);

  // 6) o + g -> per-head LN -> silu -> bf16
  post_kernel<<<BT * cH, 256, 0, stream>>>(obuf, gbuf, lnw, lnb, xbf);

  // 7) output projection
  gemm_kernel<false><<<dim3(cHID / 128, BT / 128), 256, 0, stream>>>(xbf, WoT, (float*)d_out, BT, cHID, cVD);
}
